// MemoryModule_61735859912756
// MI455X (gfx1250) — matrix operand order repaired and checked by reading, not compiled, not run on hardware
//
#include <hip/hip_runtime.h>

typedef __attribute__((ext_vector_type(16))) _Float16 v16h;
typedef __attribute__((ext_vector_type(8)))  _Float16 v8h;
typedef __attribute__((ext_vector_type(4)))  _Float16 v4h;
typedef __attribute__((ext_vector_type(8)))  float    v8f;
typedef __attribute__((ext_vector_type(2)))  float    v2f;

constexpr int kC   = 128;      // channels
constexpr int kM   = 64;       // memory bank entries
constexpr int kHW  = 128 * 128;
constexpr int kB   = 16;
constexpr int kPT  = 64;       // positions per block (4 waves x 16)
constexpr int kLDX = 136;      // f16 row stride, 272 B -> 4-bank shift per row
constexpr int kLDS = 136;

union U16 { v16h v; v8h h[2]; };
union UF8 { v8f v; float f[8]; };
union UH8 { v8h v; _Float16 h[8]; };

// f16 weight scratch in d_ws (element offsets):
//   q_w_h   [128][128] @ 0        out_w_h [128][128] @ 16384
//   mem_k_h [ 64][128] @ 32768    mem_vT_h[128][ 64] @ 40960
__global__ void prep_weights(const float* __restrict__ qw, const float* __restrict__ ow,
                             const float* __restrict__ mk, const float* __restrict__ mv,
                             _Float16* __restrict__ wsh) {
  int i = blockIdx.x * blockDim.x + threadIdx.x;
  if (i < 16384) {
    wsh[i]         = (_Float16)qw[i];
    wsh[16384 + i] = (_Float16)ow[i];
  }
  if (i < 8192) {
    wsh[32768 + i] = (_Float16)mk[i];
    int d = i >> 6, m = i & 63;                 // mem_vT[d][m] = mem_v[m][d]
    wsh[40960 + i] = (_Float16)mv[m * kC + d];
  }
}

__device__ __forceinline__ v8f wmma_f16(v16h a, v16h b, v8f c) {
  return __builtin_amdgcn_wmma_f32_16x16x32_f16(false, a, false, b, (short)0, c,
                                                false, false);
}

// A-fragment (16x32 f16) from row-major weight W[row][k].
// Lane L<16: row L, K = kc*32 + {0..7,16..23}; lane L+16: same row, K += 8.
__device__ __forceinline__ v16h gA(const _Float16* __restrict__ w, int row, int ld,
                                   int kc, int hf) {
  const _Float16* p = w + row * ld + kc * 32 + hf * 8;
  U16 u;
  u.h[0] = *(const v8h*)(p);
  u.h[1] = *(const v8h*)(p + 16);
  return u.v;
}

// B-fragment (32x16 f16) from row-major f16 LDS tile T[n][k] (B[k][n] = T[n][k]).
// Lane n<16: col n, K = kc*32 + {0..7,16..23}; lane n+16: col n, K += 8.
__device__ __forceinline__ v16h ldsB(const _Float16* base, int row, int ld,
                                     int kc, int hf) {
  const _Float16* p = base + row * ld + kc * 32 + hf * 8;
  U16 u;
  u.h[0] = *(const v8h*)(p);
  u.h[1] = *(const v8h*)(p + 16);
  return u.v;
}

__global__ __launch_bounds__(128) void fused_mem_attn(
    const float* __restrict__ x, const float* __restrict__ q_b,
    const float* __restrict__ out_b, const _Float16* __restrict__ wsh,
    float* __restrict__ out) {
  __shared__ alignas(16) _Float16 xT[kPT][kLDX];       // (position, channel) f16
  __shared__ alignas(16) _Float16 stage[4][16][kLDS];  // per-wave (position, chan)

  const int tid  = threadIdx.x;
  const int lane = tid & 31;
  const int wave = tid >> 5;
  const int hf   = lane >> 4;   // lane half: K-offset (A/B) / row-half (C/D)
  const int n    = lane & 15;   // B column = position; A row index

  const int blocksPerBatch = kHW / kPT;                // 256
  const int b  = blockIdx.x / blocksPerBatch;
  const int p0 = (blockIdx.x % blocksPerBatch) * kPT;

  const _Float16* qw_h  = wsh;
  const _Float16* ow_h  = wsh + 16384;
  const _Float16* mk_h  = wsh + 32768;
  const _Float16* mvt_h = wsh + 40960;

  // ---- cooperative fill: x[b, :, p0:p0+64] -> xT f16 (transposed) ----------
  // Each iter: 4 coalesced float2 loads (4 channels x 2 positions) ->
  // two packed v4h ds_store_b64 (4 consecutive channels at one position).
  const float* xb = x + (size_t)b * kC * kHW + p0;
#pragma unroll
  for (int it = 0; it < 8; ++it) {
    int q = tid + 128 * it;
    int c = (q >> 5) * 4;
    int p = (q & 31) * 2;
    v2f f0 = *(const v2f*)(xb + (size_t)(c + 0) * kHW + p);
    v2f f1 = *(const v2f*)(xb + (size_t)(c + 1) * kHW + p);
    v2f f2 = *(const v2f*)(xb + (size_t)(c + 2) * kHW + p);
    v2f f3 = *(const v2f*)(xb + (size_t)(c + 3) * kHW + p);
    v4h lo = { (_Float16)f0.x, (_Float16)f1.x, (_Float16)f2.x, (_Float16)f3.x };
    v4h hi = { (_Float16)f0.y, (_Float16)f1.y, (_Float16)f2.y, (_Float16)f3.y };
    *(v4h*)(&xT[p][c])     = lo;
    *(v4h*)(&xT[p + 1][c]) = hi;
  }
  __syncthreads();

  const _Float16* myX = &xT[wave * 16][0];
  _Float16*       myS = &stage[wave][0][0];

  // ============ GEMM1: Q[d][p] = q_w * x + q_b  (K = 128) ====================
  v16h bx0 = ldsB(myX, n, kLDX, 0, hf);
  v16h bx1 = ldsB(myX, n, kLDX, 1, hf);
  v16h bx2 = ldsB(myX, n, kLDX, 2, hf);
  v16h bx3 = ldsB(myX, n, kLDX, 3, hf);
#pragma unroll
  for (int t = 0; t < 8; ++t) {
    int d0 = 16 * t;
    v8f acc = {};
    acc = wmma_f16(gA(qw_h, d0 + n, kC, 0, hf), bx0, acc);
    acc = wmma_f16(gA(qw_h, d0 + n, kC, 1, hf), bx1, acc);
    acc = wmma_f16(gA(qw_h, d0 + n, kC, 2, hf), bx2, acc);
    acc = wmma_f16(gA(qw_h, d0 + n, kC, 3, hf), bx3, acc);
    UF8 u;  u.v  = acc;
    UF8 qb; qb.v = *(const v8f*)(q_b + d0 + 8 * hf);   // d = d0 + 8*hf + v
    v4h s0 = { (_Float16)(u.f[0] + qb.f[0]), (_Float16)(u.f[1] + qb.f[1]),
               (_Float16)(u.f[2] + qb.f[2]), (_Float16)(u.f[3] + qb.f[3]) };
    v4h s1 = { (_Float16)(u.f[4] + qb.f[4]), (_Float16)(u.f[5] + qb.f[5]),
               (_Float16)(u.f[6] + qb.f[6]), (_Float16)(u.f[7] + qb.f[7]) };
    *(v4h*)(myS + n * kLDS + d0 + 8 * hf)     = s0;    // stage[p][d], packed x4
    *(v4h*)(myS + n * kLDS + d0 + 8 * hf + 4) = s1;
  }

  // ============ GEMM2: logits[m][p] = mem_k * Q  (K = 128) ===================
  v16h bq0 = ldsB(myS, n, kLDS, 0, hf);
  v16h bq1 = ldsB(myS, n, kLDS, 1, hf);
  v16h bq2 = ldsB(myS, n, kLDS, 2, hf);
  v16h bq3 = ldsB(myS, n, kLDS, 3, hf);
  UF8 lg[4];
#pragma unroll
  for (int t = 0; t < 4; ++t) {
    v8f acc = {};
    acc = wmma_f16(gA(mk_h, 16 * t + n, kC, 0, hf), bq0, acc);
    acc = wmma_f16(gA(mk_h, 16 * t + n, kC, 1, hf), bq1, acc);
    acc = wmma_f16(gA(mk_h, 16 * t + n, kC, 2, hf), bq2, acc);
    acc = wmma_f16(gA(mk_h, 16 * t + n, kC, 3, hf), bq3, acc);
    lg[t].v = acc;
  }

  // ==== softmax over m=64 for this lane's position: lane pair (l, l^16) =====
  float mx = lg[0].f[0];
#pragma unroll
  for (int t = 0; t < 4; ++t)
#pragma unroll
    for (int v = 0; v < 8; ++v) mx = fmaxf(mx, lg[t].f[v]);
  mx = fmaxf(mx, __shfl_xor(mx, 16, 32));
  float sum = 0.f;
#pragma unroll
  for (int t = 0; t < 4; ++t)
#pragma unroll
    for (int v = 0; v < 8; ++v) { lg[t].f[v] = __expf(lg[t].f[v] - mx); sum += lg[t].f[v]; }
  sum += __shfl_xor(sum, 16, 32);
  float inv = 1.0f / sum;
#pragma unroll
  for (int t = 0; t < 4; ++t) {           // stage[p][m] packed x4, m = 16t+8hf+v
    v4h s0 = { (_Float16)(lg[t].f[0] * inv), (_Float16)(lg[t].f[1] * inv),
               (_Float16)(lg[t].f[2] * inv), (_Float16)(lg[t].f[3] * inv) };
    v4h s1 = { (_Float16)(lg[t].f[4] * inv), (_Float16)(lg[t].f[5] * inv),
               (_Float16)(lg[t].f[6] * inv), (_Float16)(lg[t].f[7] * inv) };
    *(v4h*)(myS + n * kLDS + 16 * t + 8 * hf)     = s0;
    *(v4h*)(myS + n * kLDS + 16 * t + 8 * hf + 4) = s1;
  }

  // ====== GEMM3: mem_out[d][p] = mem_v^T * attn (K = 64) + residual x =======
  v16h ba0 = ldsB(myS, n, kLDS, 0, hf);
  v16h ba1 = ldsB(myS, n, kLDS, 1, hf);
#pragma unroll
  for (int t = 0; t < 8; ++t) {
    int d0 = 16 * t;
    v8f acc = {};
    acc = wmma_f16(gA(mvt_h, d0 + n, kM, 0, hf), ba0, acc);
    acc = wmma_f16(gA(mvt_h, d0 + n, kM, 1, hf), ba1, acc);
    UF8 u; u.v = acc;
    UH8 xr; xr.v = *(const v8h*)(&xT[wave * 16 + n][d0 + 8 * hf]);  // x residual
    v4h s0 = { (_Float16)(u.f[0] + (float)xr.h[0]), (_Float16)(u.f[1] + (float)xr.h[1]),
               (_Float16)(u.f[2] + (float)xr.h[2]), (_Float16)(u.f[3] + (float)xr.h[3]) };
    v4h s1 = { (_Float16)(u.f[4] + (float)xr.h[4]), (_Float16)(u.f[5] + (float)xr.h[5]),
               (_Float16)(u.f[6] + (float)xr.h[6]), (_Float16)(u.f[7] + (float)xr.h[7]) };
    *(v4h*)(myS + n * kLDS + d0 + 8 * hf)     = s0;
    *(v4h*)(myS + n * kLDS + d0 + 8 * hf + 4) = s1;
  }

  // ====== GEMM4: out[e][p] = out_w * (x + mem_out) + out_b (K = 128) ========
  v16h bo0 = ldsB(myS, n, kLDS, 0, hf);
  v16h bo1 = ldsB(myS, n, kLDS, 1, hf);
  v16h bo2 = ldsB(myS, n, kLDS, 2, hf);
  v16h bo3 = ldsB(myS, n, kLDS, 3, hf);
  float* pout = out + (size_t)b * kC * kHW + p0 + wave * 16 + n;  // this lane's p
#pragma unroll
  for (int t = 0; t < 8; ++t) {
    int e0 = 16 * t;
    v8f acc = {};
    acc = wmma_f16(gA(ow_h, e0 + n, kC, 0, hf), bo0, acc);
    acc = wmma_f16(gA(ow_h, e0 + n, kC, 1, hf), bo1, acc);
    acc = wmma_f16(gA(ow_h, e0 + n, kC, 2, hf), bo2, acc);
    acc = wmma_f16(gA(ow_h, e0 + n, kC, 3, hf), bo3, acc);
    UF8 u;  u.v  = acc;
    UF8 ob; ob.v = *(const v8f*)(out_b + e0 + 8 * hf);
#pragma unroll
    for (int v = 0; v < 8; ++v)   // coalesced: 16 consecutive p across lanes
      pout[(size_t)(e0 + v + 8 * hf) * kHW] = u.f[v] + ob.f[v];
  }
}

extern "C" void kernel_launch(void* const* d_in, const int* in_sizes, int n_in,
                              void* d_out, int out_size, void* d_ws, size_t ws_size,
                              hipStream_t stream) {
  const float* x  = (const float*)d_in[0];
  const float* qw = (const float*)d_in[1];
  const float* qb = (const float*)d_in[2];
  const float* ow = (const float*)d_in[3];
  const float* ob = (const float*)d_in[4];
  const float* mk = (const float*)d_in[5];
  const float* mv = (const float*)d_in[6];
  _Float16* wsh = (_Float16*)d_ws;   // 96 KiB of scratch

  prep_weights<<<64, 256, 0, stream>>>(qw, ow, mk, mv, wsh);

  int nblocks = kB * (kHW / kPT);    // 4096 blocks x 128 threads (4 waves)
  fused_mem_attn<<<nblocks, 128, 0, stream>>>(x, qb, ob, wsh, (float*)d_out);
}
